// QuantizedLinearINT4_41996190220298
// MI455X (gfx1250) — matrix operand order repaired and checked by reading, not compiled, not run on hardware
//
#include <hip/hip_runtime.h>

typedef __attribute__((ext_vector_type(16))) __bf16        v16bf;
typedef __attribute__((ext_vector_type(8)))  float         v8f;
typedef __attribute__((ext_vector_type(4)))  unsigned int  v4u;
typedef __attribute__((ext_vector_type(2)))  unsigned int  v2u;

#define BM 128
#define BN 128
#define BK 64
#define LDT 72   // padded LDS row stride (halves): 144B rows -> 16B-aligned chunks

union FragU { v16bf v; v4u q[2]; };

__device__ __forceinline__ unsigned int pack_bf16x2(float a, float b) {
  // round-to-nearest-even f32 -> bf16, packed low|high
  unsigned int ua = __float_as_uint(a);
  unsigned int ub = __float_as_uint(b);
  ua += 0x7fffu + ((ua >> 16) & 1u);
  ub += 0x7fffu + ((ub >> 16) & 1u);
  return (ua >> 16) | (ub & 0xffff0000u);
}

// ---------------------------------------------------------------------------
// Phase 1a: X (f32) -> Xbf (bf16), one pass. 4 floats / thread.
// ---------------------------------------------------------------------------
__global__ __launch_bounds__(256)
void cvt_x_bf16(const float* __restrict__ X, unsigned short* __restrict__ Xb) {
  const size_t t = (size_t)blockIdx.x * 256 + threadIdx.x;
  float4 x = *reinterpret_cast<const float4*>(X + t * 4);
  v2u p;
  p.x = pack_bf16x2(x.x, x.y);
  p.y = pack_bf16x2(x.z, x.w);
  *reinterpret_cast<v2u*>(Xb + t * 4) = p;
}

// ---------------------------------------------------------------------------
// Phase 1b: int4 qweight -> Wbf (bf16), one pass.
// One thread: 4 int32 (= 4 bytes = 8 weights, one quant group slice).
// ---------------------------------------------------------------------------
__global__ __launch_bounds__(256)
void dequant_w_bf16(const int* __restrict__ Q, const float* __restrict__ S,
                    const float* __restrict__ Z, unsigned short* __restrict__ Wb,
                    int K, int G) {
  const int t = blockIdx.x * 256 + threadIdx.x;
  const int cpr = K >> 3;                 // int4 chunks per output row
  const int row = t / cpr;
  const int c   = t - row * cpr;
  const int gs  = K / G;
  const int g   = (c * 8) / gs;
  const float s  = S[(size_t)row * G + g];
  const float cz = -(8.0f + Z[(size_t)row * G + g]) * s;   // w = q*s + cz
  const int4 w = *reinterpret_cast<const int4*>(Q + (size_t)t * 4);
  const int b4[4] = { w.x, w.y, w.z, w.w };
  v4u p;
#pragma unroll
  for (int e = 0; e < 4; ++e) {
    const unsigned int bb = ((unsigned int)b4[e]) & 0xffu;
    const float lo = fmaf((float)(bb & 15u), s, cz);       // even k
    const float hi = fmaf((float)(bb >> 4),  s, cz);       // odd k
    p[e] = pack_bf16x2(lo, hi);
  }
  *reinterpret_cast<v4u*>(Wb + (size_t)t * 8) = p;
}

// ---------------------------------------------------------------------------
// Phase 2: pure bf16 GEMM, 128x128 block tile, 8 wave32, 64x32 wave tile.
// Inner loop is b128 loads + v_wmma only (near-zero VALU).
// ---------------------------------------------------------------------------
__global__ __launch_bounds__(256)
void bf16_gemm_wmma(const unsigned short* __restrict__ Xb,
                    const unsigned short* __restrict__ Wb,
                    const float* __restrict__ Bias, float* __restrict__ Y,
                    int M, int N, int K)
{
  __shared__ alignas(16) unsigned short ldsX[BM * LDT];
  __shared__ alignas(16) unsigned short ldsW[BN * LDT];

  const int tid  = threadIdx.x;
  const int lane = tid & 31;
  const int wv   = tid >> 5;
  const int wm   = (wv & 1) * 64;
  const int wn   = (wv >> 1) * 32;
  const int lr   = lane & 15;
  const int lh   = lane >> 4;

  const int m0 = blockIdx.y * BM;
  const int n0 = blockIdx.x * BN;

  v8f acc[4][2] = {};

  const int ktiles = K / BK;
  for (int kt = 0; kt < ktiles; ++kt) {
    const int kbase = kt * BK;

    // global loads: 4 b128 of X + 4 b128 of W per thread (coalesced)
    v4u xr[4], wr[4];
#pragma unroll
    for (int j = 0; j < 4; ++j) {
      const int f   = tid + 256 * j;      // b128 id: 8 per 64-wide row
      const int row = f >> 3;
      const int col = (f & 7) << 3;
      xr[j] = *reinterpret_cast<const v4u*>(Xb + (size_t)(m0 + row) * K + kbase + col);
      wr[j] = *reinterpret_cast<const v4u*>(Wb + (size_t)(n0 + row) * K + kbase + col);
    }

    __syncthreads();                      // previous slice consumed

#pragma unroll
    for (int j = 0; j < 4; ++j) {
      const int f   = tid + 256 * j;
      const int row = f >> 3;
      const int col = (f & 7) << 3;
      *reinterpret_cast<v4u*>(&ldsX[row * LDT + col]) = xr[j];
      *reinterpret_cast<v4u*>(&ldsW[row * LDT + col]) = wr[j];
    }

    __syncthreads();                      // tile visible

#pragma unroll
    for (int kk = 0; kk < BK; kk += 32) {
      FragU a[4], b[2];
      // A layout: lane-half 0 -> K 0-7 & 16-23 ; half 1 -> K 8-15 & 24-31
      const int ka = kk + (lh << 3);
#pragma unroll
      for (int im = 0; im < 4; ++im) {
        const unsigned short* p = &ldsX[(wm + im * 16 + lr) * LDT + ka];
        a[im].q[0] = *reinterpret_cast<const v4u*>(p);
        a[im].q[1] = *reinterpret_cast<const v4u*>(p + 16);
      }
      // B layout: lane-half 0 -> K 0-7 & 16-23 ; half 1 -> K 8-15 & 24-31
      const int kb = kk + (lh << 3);
#pragma unroll
      for (int in = 0; in < 2; ++in) {
        const unsigned short* p = &ldsW[(wn + in * 16 + lr) * LDT + kb];
        b[in].q[0] = *reinterpret_cast<const v4u*>(p);
        b[in].q[1] = *reinterpret_cast<const v4u*>(p + 16);
      }
#pragma unroll
      for (int im = 0; im < 4; ++im)
#pragma unroll
        for (int in = 0; in < 2; ++in)
          acc[im][in] = __builtin_amdgcn_wmma_f32_16x16x32_bf16(
              false, a[im].v, false, b[in].v,
              (short)0, acc[im][in], false, false);
    }
  }

#pragma unroll
  for (int in = 0; in < 2; ++in) {
    const int nc = n0 + wn + in * 16 + lr;
    const float bv = Bias[nc];
#pragma unroll
    for (int im = 0; im < 4; ++im) {
      const int mr = m0 + wm + im * 16 + lh * 8;
      float* yp = Y + (size_t)mr * N + nc;
#pragma unroll
      for (int i = 0; i < 8; ++i)
        yp[(size_t)i * N] = acc[im][in][i] + bv;
    }
  }
}

// ---------------------------------------------------------------------------
// Fallback: fully fused kernel (used when d_ws is too small). Same as round 1.
// ---------------------------------------------------------------------------
__global__ __launch_bounds__(256)
void q4_gemm_wmma(const float* __restrict__ X, const int* __restrict__ Q,
                  const float* __restrict__ S, const float* __restrict__ Z,
                  const float* __restrict__ Bias, float* __restrict__ Y,
                  int M, int N, int K, int G)
{
  __shared__ alignas(16) unsigned short ldsX[BM * LDT];
  __shared__ alignas(16) unsigned short ldsW[BN * LDT];

  const int tid  = threadIdx.x;
  const int lane = tid & 31;
  const int wv   = tid >> 5;
  const int wm   = (wv & 1) * 64;
  const int wn   = (wv >> 1) * 32;
  const int lr   = lane & 15;
  const int lh   = lane >> 4;

  const int m0 = blockIdx.y * BM;
  const int n0 = blockIdx.x * BN;
  const int gs = K / G;
  const int kbytes = K >> 1;

  v8f acc[4][2] = {};

  const int ktiles = K / BK;
  for (int kt = 0; kt < ktiles; ++kt) {
    const int kbase = kt * BK;
    float4 xr[8];
#pragma unroll
    for (int j = 0; j < 8; ++j) {
      const int f   = tid + 256 * j;
      const int row = f >> 4;
      const int col = (f & 15) << 2;
      xr[j] = *reinterpret_cast<const float4*>(X + (size_t)(m0 + row) * K + (kbase + col));
    }
    int4 wr[4]; float sc[4], cz[4];
    const int g = kbase / gs;
#pragma unroll
    for (int j = 0; j < 4; ++j) {
      const int q   = tid + 256 * j;
      const int row = q >> 3;
      const int kc  = q & 7;
      wr[j] = *reinterpret_cast<const int4*>(Q + (size_t)(n0 + row) * kbytes + (kbase >> 1) + kc * 4);
      const float s = S[(size_t)(n0 + row) * G + g];
      const float z = Z[(size_t)(n0 + row) * G + g];
      sc[j] = s;
      cz[j] = -(8.0f + z) * s;
    }

    __syncthreads();

#pragma unroll
    for (int j = 0; j < 8; ++j) {
      const int f   = tid + 256 * j;
      const int row = f >> 4;
      const int col = (f & 15) << 2;
      v2u p;
      p.x = pack_bf16x2(xr[j].x, xr[j].y);
      p.y = pack_bf16x2(xr[j].z, xr[j].w);
      *reinterpret_cast<v2u*>(&ldsX[row * LDT + col]) = p;
    }
#pragma unroll
    for (int j = 0; j < 4; ++j) {
      const int q   = tid + 256 * j;
      const int row = q >> 3;
      const int kc  = q & 7;
      const int b4[4] = { wr[j].x, wr[j].y, wr[j].z, wr[j].w };
      v4u p;
#pragma unroll
      for (int e = 0; e < 4; ++e) {
        const unsigned int bb = ((unsigned int)b4[e]) & 0xffu;
        const float lo = fmaf((float)(bb & 15u), sc[j], cz[j]);
        const float hi = fmaf((float)(bb >> 4),  sc[j], cz[j]);
        p[e] = pack_bf16x2(lo, hi);
      }
      *reinterpret_cast<v4u*>(&ldsW[row * LDT + kc * 8]) = p;
    }

    __syncthreads();

#pragma unroll
    for (int kk = 0; kk < BK; kk += 32) {
      FragU a[4], b[2];
      const int ka = kk + (lh << 3);
#pragma unroll
      for (int im = 0; im < 4; ++im) {
        const unsigned short* p = &ldsX[(wm + im * 16 + lr) * LDT + ka];
        a[im].q[0] = *reinterpret_cast<const v4u*>(p);
        a[im].q[1] = *reinterpret_cast<const v4u*>(p + 16);
      }
      const int kb = kk + (lh << 3);
#pragma unroll
      for (int in = 0; in < 2; ++in) {
        const unsigned short* p = &ldsW[(wn + in * 16 + lr) * LDT + kb];
        b[in].q[0] = *reinterpret_cast<const v4u*>(p);
        b[in].q[1] = *reinterpret_cast<const v4u*>(p + 16);
      }
#pragma unroll
      for (int im = 0; im < 4; ++im)
#pragma unroll
        for (int in = 0; in < 2; ++in)
          acc[im][in] = __builtin_amdgcn_wmma_f32_16x16x32_bf16(
              false, a[im].v, false, b[in].v,
              (short)0, acc[im][in], false, false);
    }
  }

#pragma unroll
  for (int in = 0; in < 2; ++in) {
    const int nc = n0 + wn + in * 16 + lr;
    const float bv = Bias[nc];
#pragma unroll
    for (int im = 0; im < 4; ++im) {
      const int mr = m0 + wm + im * 16 + lh * 8;
      float* yp = Y + (size_t)mr * N + nc;
#pragma unroll
      for (int i = 0; i < 8; ++i)
        yp[(size_t)i * N] = acc[im][in][i] + bv;
    }
  }
}

extern "C" void kernel_launch(void* const* d_in, const int* in_sizes, int n_in,
                              void* d_out, int out_size, void* d_ws, size_t ws_size,
                              hipStream_t stream) {
  const float* X    = (const float*)d_in[0];
  const int*   Q    = (const int*)d_in[1];
  const float* S    = (const float*)d_in[2];
  const float* Z    = (const float*)d_in[3];
  const float* Bias = (const float*)d_in[4];
  float*       Y    = (float*)d_out;

  const int O = in_sizes[4];                         // 11008
  const int K = (int)((2LL * in_sizes[1]) / O);      // 4096
  const int M = (int)(in_sizes[0] / K);              // 8192
  const int G = in_sizes[2] / O;                     // 32

  const size_t bytesX = (size_t)M * K * 2;           // bf16 X
  const size_t bytesW = (size_t)O * K * 2;           // bf16 W
  dim3 grid(O / BN, M / BM);                         // 86 x 64

  if (ws_size >= bytesX + bytesW) {
    unsigned short* Xb = (unsigned short*)d_ws;
    unsigned short* Wb = Xb + (size_t)M * K;

    const int nblkX = (int)(((size_t)M * K / 4) / 256);      // 4 f32 / thread
    cvt_x_bf16<<<nblkX, 256, 0, stream>>>(X, Xb);

    const int nblkW = (int)(((size_t)O * (K >> 3)) / 256);   // 8 weights / thread
    dequant_w_bf16<<<nblkW, 256, 0, stream>>>(Q, S, Z, Wb, K, G);

    bf16_gemm_wmma<<<grid, 256, 0, stream>>>(Xb, Wb, Bias, Y, M, O, K);
  } else {
    q4_gemm_wmma<<<grid, 256, 0, stream>>>(X, Q, S, Z, Bias, Y, M, O, K, G);
  }
}
